// NodeModel_49246095016467
// MI455X (gfx1250) — matrix operand order repaired and checked by reading, not compiled, not run on hardware
//
#include <hip/hip_runtime.h>

// ---------------- problem constants (from reference) ----------------
#define N_NODES 100000
#define N_EDGES 1600000
#define D_NODE  128
#define D_EDGE  128
#define D_GLOB  64
#define D_HID   128
// edge MLP:  K1 = 256 (x||edge_attr), node MLP: K2 = 320 (x||agg||u)

typedef __attribute__((ext_vector_type(16))) __bf16 v16bf;
typedef __attribute__((ext_vector_type(8)))  float  v8f;
typedef __attribute__((ext_vector_type(4)))  unsigned int u32x4;

union AFrag { v16bf v; u32x4 q[2]; __bf16 h[16]; };

// -------------------------------------------------------------------
// Pack a KxN fp32 weight matrix into bf16 "B-fragment" tile order for
// v_wmma_f32_16x16x32_bf16:  tile T = kt*(N/16)+nt holds 32 lanes x 16
// bf16 contiguous; element (lane, v*2+p) = W[kt*32 + (v<4?0:16) + (lane>=16?8:0) +
// 2(v&3) + p][nt*16 + lane&15].  One 32B load per lane fetches a fragment.
// -------------------------------------------------------------------
__global__ void pack_w_kernel(const float* __restrict__ W, __bf16* __restrict__ out,
                              int K, int N) {
    int idx = blockIdx.x * 256 + threadIdx.x;
    if (idx >= K * N) return;
    int p    = idx & 1;
    int v    = (idx >> 1) & 7;
    int lane = (idx >> 4) & 31;
    int tile = idx >> 9;
    int ntn  = N >> 4;
    int nt   = tile % ntn;
    int kt   = tile / ntn;
    int n = nt * 16 + (lane & 15);
    int k = kt * 32 + ((v >> 2) << 4) + ((lane >> 4) << 3) + (v & 3) * 2 + p;   // B layout: K = (v<4?0:16) + g*8 + 2(v&3) + p
    out[idx] = (__bf16)W[k * N + n];
}

__global__ void cvt_f32_bf16_kernel(const float* __restrict__ in,
                                    __bf16* __restrict__ out, int n) {
    int idx = blockIdx.x * 256 + threadIdx.x;
    if (idx < n) out[idx] = (__bf16)in[idx];
}

__global__ void zero_f32_kernel(float* __restrict__ a, int n) {
    int idx = blockIdx.x * 256 + threadIdx.x;
    if (idx < n) a[idx] = 0.0f;
}

__device__ __forceinline__ void cvt8(const float* __restrict__ src, __bf16* dst) {
    float4 lo = *(const float4*)src;
    float4 hi = *(const float4*)(src + 4);
    dst[0] = (__bf16)lo.x; dst[1] = (__bf16)lo.y; dst[2] = (__bf16)lo.z; dst[3] = (__bf16)lo.w;
    dst[4] = (__bf16)hi.x; dst[5] = (__bf16)hi.y; dst[6] = (__bf16)hi.z; dst[7] = (__bf16)hi.w;
}
__device__ __forceinline__ void cvt8s(const float* __restrict__ src, float s, __bf16* dst) {
    float4 lo = *(const float4*)src;
    float4 hi = *(const float4*)(src + 4);
    dst[0] = (__bf16)(lo.x*s); dst[1] = (__bf16)(lo.y*s); dst[2] = (__bf16)(lo.z*s); dst[3] = (__bf16)(lo.w*s);
    dst[4] = (__bf16)(hi.x*s); dst[5] = (__bf16)(hi.y*s); dst[6] = (__bf16)(hi.z*s); dst[7] = (__bf16)(hi.w*s);
}

// -------------------------------------------------------------------
// Edge MLP + scatter-add.  One wave per 16-edge tile.
//   GEMM1: A(16x256 bf16, gathered) x W1a(256x128)  [8 kt x 8 nt wmma]
//   bias+ReLU -> bf16 -> per-wave LDS slice
//   GEMM2: h(16x128) x W1b(128x128)                  [4 kt x 8 nt wmma]
//   bias -> atomic scatter into summed[col], cnt[col]
// -------------------------------------------------------------------
__global__ __launch_bounds__(256) void edge_mlp_kernel(
    const int* __restrict__ ei, const float* __restrict__ eattr,
    const __bf16* __restrict__ xbf,
    const __bf16* __restrict__ w1aP, const __bf16* __restrict__ w1bP,
    const float* __restrict__ b1a, const float* __restrict__ b1b,
    float* __restrict__ summed, float* __restrict__ cnt, int numTiles)
{
    __shared__ __bf16 sWa[256 * 128];       // 64 KB packed W1a
    __shared__ __bf16 sWb[128 * 128];       // 32 KB packed W1b
    __shared__ __bf16 sH[8][16 * D_HID];    // 32 KB per-wave h staging

    for (int i = threadIdx.x; i < 256 * 128 / 8; i += 256)
        ((u32x4*)sWa)[i] = ((const u32x4*)w1aP)[i];
    for (int i = threadIdx.x; i < 128 * 128 / 8; i += 256)
        ((u32x4*)sWb)[i] = ((const u32x4*)w1bP)[i];
    __syncthreads();

    const int wave = threadIdx.x >> 5;
    const int lane = threadIdx.x & 31;
    const int g    = lane >> 4;      // lane group (A: +8 K, B: +8 K)
    const int mrow = lane & 15;      // my A-matrix row
    __bf16* h = sH[wave];
    const v8f vzero = {0.f,0.f,0.f,0.f,0.f,0.f,0.f,0.f};

    for (int tile = blockIdx.x * 8 + wave; tile < numTiles; tile += gridDim.x * 8) {
        const int ebase = tile * 16;
        const int r = ei[ebase + mrow];                       // source node of my row
        const __bf16* xrow = xbf   + (size_t)r * D_NODE;
        const float*  arow = eattr + (size_t)(ebase + mrow) * D_EDGE;

        v8f acc[8];
        #pragma unroll
        for (int nt = 0; nt < 8; nt++) acc[nt] = vzero;

        #pragma unroll
        for (int kt = 0; kt < 8; kt++) {
            AFrag a;
            const int kb = kt * 32 + g * 8;                   // A layout: K = (v<4?0:16)+g*8+2(v&3)+p
            if (kt < 4) {                                     // x part, already bf16
                a.q[0] = *(const u32x4*)(xrow + kb);
                a.q[1] = *(const u32x4*)(xrow + kb + 16);
            } else {                                          // edge_attr part, cvt f32->bf16
                const int c0 = kb - 128;
                cvt8(arow + c0,      a.h);
                cvt8(arow + c0 + 16, a.h + 8);
            }
            #pragma unroll
            for (int nt = 0; nt < 8; nt++) {
                AFrag b;
                const __bf16* bp = sWa + (kt * 8 + nt) * 512 + lane * 16;
                b.q[0] = *(const u32x4*)bp;
                b.q[1] = *(const u32x4*)(bp + 8);
                acc[nt] = __builtin_amdgcn_wmma_f32_16x16x32_bf16(
                    false, a.v, false, b.v, (short)0, acc[nt], false, false);
            }
        }

        // bias + ReLU, round-trip through LDS to re-stripe M-major -> K-major
        #pragma unroll
        for (int nt = 0; nt < 8; nt++) {
            const int n = nt * 16 + mrow;
            const float bias = b1a[n];
            #pragma unroll
            for (int vv = 0; vv < 8; vv++) {
                float c = acc[nt][vv] + bias;
                c = fmaxf(c, 0.0f);
                h[(g * 8 + vv) * D_HID + n] = (__bf16)c;      // C layout: row = g*8+v, col = n
            }
        }
        asm volatile("s_wait_dscnt 0x0" ::: "memory");        // same-wave DS RAW drain

        #pragma unroll
        for (int nt = 0; nt < 8; nt++) acc[nt] = vzero;
        #pragma unroll
        for (int kt = 0; kt < 4; kt++) {
            AFrag a;
            const __bf16* ap = h + mrow * D_HID + kt * 32 + g * 8;
            a.q[0] = *(const u32x4*)ap;
            a.q[1] = *(const u32x4*)(ap + 16);
            #pragma unroll
            for (int nt = 0; nt < 8; nt++) {
                AFrag b;
                const __bf16* bp = sWb + (kt * 8 + nt) * 512 + lane * 16;
                b.q[0] = *(const u32x4*)bp;
                b.q[1] = *(const u32x4*)(bp + 8);
                acc[nt] = __builtin_amdgcn_wmma_f32_16x16x32_bf16(
                    false, a.v, false, b.v, (short)0, acc[nt], false, false);
            }
        }

        int cm[8];
        #pragma unroll
        for (int vv = 0; vv < 8; vv++) cm[vv] = ei[N_EDGES + ebase + g * 8 + vv];

        #pragma unroll
        for (int nt = 0; nt < 8; nt++) {
            const int n = nt * 16 + mrow;
            const float bias = b1b[n];
            #pragma unroll
            for (int vv = 0; vv < 8; vv++) {
                const float val = acc[nt][vv] + bias;
                unsafeAtomicAdd(&summed[(size_t)cm[vv] * D_HID + n], val);  // L2-resident fp32 atomic
            }
        }
        if (lane < 16)
            unsafeAtomicAdd(&cnt[ei[N_EDGES + ebase + lane]], 1.0f);
    }
}

// -------------------------------------------------------------------
// Node MLP.  One wave per 16-node tile; K = 320 = x(128)||agg(128)||u(64)
// -------------------------------------------------------------------
__global__ __launch_bounds__(256) void node_mlp_kernel(
    const __bf16* __restrict__ xbf, const float* __restrict__ u,
    const int* __restrict__ batch,
    const __bf16* __restrict__ w2aP, const __bf16* __restrict__ w2bP,
    const float* __restrict__ b2a, const float* __restrict__ b2b,
    const float* __restrict__ summed, const float* __restrict__ cnt,
    float* __restrict__ out, int numTiles)
{
    __shared__ __bf16 sWa[320 * 128];       // 80 KB packed W2a
    __shared__ __bf16 sWb[128 * 128];       // 32 KB packed W2b
    __shared__ __bf16 sH[8][16 * D_HID];    // 32 KB

    for (int i = threadIdx.x; i < 320 * 128 / 8; i += 256)
        ((u32x4*)sWa)[i] = ((const u32x4*)w2aP)[i];
    for (int i = threadIdx.x; i < 128 * 128 / 8; i += 256)
        ((u32x4*)sWb)[i] = ((const u32x4*)w2bP)[i];
    __syncthreads();

    const int wave = threadIdx.x >> 5;
    const int lane = threadIdx.x & 31;
    const int g    = lane >> 4;
    const int mrow = lane & 15;
    __bf16* h = sH[wave];
    const v8f vzero = {0.f,0.f,0.f,0.f,0.f,0.f,0.f,0.f};

    for (int tile = blockIdx.x * 8 + wave; tile < numTiles; tile += gridDim.x * 8) {
        const int node = tile * 16 + mrow;
        const __bf16* xrow = xbf    + (size_t)node * D_NODE;
        const float*  srow = summed + (size_t)node * D_HID;
        const float   invc = 1.0f / fmaxf(cnt[node], 1.0f);
        const float*  urow = u + (size_t)batch[node] * D_GLOB;

        v8f acc[8];
        #pragma unroll
        for (int nt = 0; nt < 8; nt++) acc[nt] = vzero;

        #pragma unroll
        for (int kt = 0; kt < 10; kt++) {
            AFrag a;
            const int kb = kt * 32 + g * 8;
            if (kt < 4) {                                     // x (bf16)
                a.q[0] = *(const u32x4*)(xrow + kb);
                a.q[1] = *(const u32x4*)(xrow + kb + 16);
            } else if (kt < 8) {                              // agg = summed * invc
                const int c0 = kb - 128;
                cvt8s(srow + c0,      invc, a.h);
                cvt8s(srow + c0 + 16, invc, a.h + 8);
            } else {                                          // u[batch]
                const int c0 = kb - 256;
                cvt8(urow + c0,      a.h);
                cvt8(urow + c0 + 16, a.h + 8);
            }
            #pragma unroll
            for (int nt = 0; nt < 8; nt++) {
                AFrag b;
                const __bf16* bp = sWa + (kt * 8 + nt) * 512 + lane * 16;
                b.q[0] = *(const u32x4*)bp;
                b.q[1] = *(const u32x4*)(bp + 8);
                acc[nt] = __builtin_amdgcn_wmma_f32_16x16x32_bf16(
                    false, a.v, false, b.v, (short)0, acc[nt], false, false);
            }
        }

        #pragma unroll
        for (int nt = 0; nt < 8; nt++) {
            const int n = nt * 16 + mrow;
            const float bias = b2a[n];
            #pragma unroll
            for (int vv = 0; vv < 8; vv++) {
                float c = acc[nt][vv] + bias;
                c = fmaxf(c, 0.0f);
                h[(g * 8 + vv) * D_HID + n] = (__bf16)c;
            }
        }
        asm volatile("s_wait_dscnt 0x0" ::: "memory");

        #pragma unroll
        for (int nt = 0; nt < 8; nt++) acc[nt] = vzero;
        #pragma unroll
        for (int kt = 0; kt < 4; kt++) {
            AFrag a;
            const __bf16* ap = h + mrow * D_HID + kt * 32 + g * 8;
            a.q[0] = *(const u32x4*)ap;
            a.q[1] = *(const u32x4*)(ap + 16);
            #pragma unroll
            for (int nt = 0; nt < 8; nt++) {
                AFrag b;
                const __bf16* bp = sWb + (kt * 8 + nt) * 512 + lane * 16;
                b.q[0] = *(const u32x4*)bp;
                b.q[1] = *(const u32x4*)(bp + 8);
                acc[nt] = __builtin_amdgcn_wmma_f32_16x16x32_bf16(
                    false, a.v, false, b.v, (short)0, acc[nt], false, false);
            }
        }

        #pragma unroll
        for (int nt = 0; nt < 8; nt++) {
            const int n = nt * 16 + mrow;
            const float bias = b2b[n];
            #pragma unroll
            for (int vv = 0; vv < 8; vv++) {
                out[(size_t)(tile * 16 + g * 8 + vv) * D_HID + n] = acc[nt][vv] + bias;
            }
        }
    }
}

// -------------------------------------------------------------------
extern "C" void kernel_launch(void* const* d_in, const int* in_sizes, int n_in,
                              void* d_out, int out_size, void* d_ws, size_t ws_size,
                              hipStream_t stream)
{
    const float* x     = (const float*)d_in[0];
    const int*   ei    = (const int*)  d_in[1];
    const float* eattr = (const float*)d_in[2];
    const float* u     = (const float*)d_in[3];
    const int*   batch = (const int*)  d_in[4];
    const float* W1a = (const float*)d_in[5];
    const float* b1a = (const float*)d_in[6];
    const float* W1b = (const float*)d_in[7];
    const float* b1b = (const float*)d_in[8];
    const float* W2a = (const float*)d_in[9];
    const float* b2a = (const float*)d_in[10];
    const float* W2b = (const float*)d_in[11];
    const float* b2b = (const float*)d_in[12];
    (void)in_sizes; (void)n_in; (void)out_size; (void)ws_size;

    char* ws = (char*)d_ws;
    size_t off = 0;
    auto take = [&](size_t bytes) {
        char* p = ws + off;
        off = (off + bytes + 255) & ~(size_t)255;
        return p;
    };
    __bf16* xbf    = (__bf16*)take((size_t)N_NODES * D_NODE * 2);
    __bf16* w1aP   = (__bf16*)take((size_t)256 * 128 * 2);
    __bf16* w1bP   = (__bf16*)take((size_t)128 * 128 * 2);
    __bf16* w2aP   = (__bf16*)take((size_t)320 * 128 * 2);
    __bf16* w2bP   = (__bf16*)take((size_t)128 * 128 * 2);
    float*  summed = (float*) take((size_t)N_NODES * D_HID * 4);
    float*  cnt    = (float*) take((size_t)N_NODES * 4);

    const int T = 256;
    cvt_f32_bf16_kernel<<<(N_NODES * D_NODE + T - 1) / T, T, 0, stream>>>(x, xbf, N_NODES * D_NODE);
    pack_w_kernel<<<(256 * 128 + T - 1) / T, T, 0, stream>>>(W1a, w1aP, 256, 128);
    pack_w_kernel<<<(128 * 128 + T - 1) / T, T, 0, stream>>>(W1b, w1bP, 128, 128);
    pack_w_kernel<<<(320 * 128 + T - 1) / T, T, 0, stream>>>(W2a, w2aP, 320, 128);
    pack_w_kernel<<<(128 * 128 + T - 1) / T, T, 0, stream>>>(W2b, w2bP, 128, 128);
    zero_f32_kernel<<<(N_NODES * D_HID + T - 1) / T, T, 0, stream>>>(summed, N_NODES * D_HID);
    zero_f32_kernel<<<(N_NODES + T - 1) / T, T, 0, stream>>>(cnt, N_NODES);

    edge_mlp_kernel<<<2048, 256, 0, stream>>>(ei, eattr, xbf, w1aP, w1bP, b1a, b1b,
                                              summed, cnt, N_EDGES / 16);
    node_mlp_kernel<<<(N_NODES / 16 + 7) / 8, 256, 0, stream>>>(
        xbf, u, batch, w2aP, w2bP, b2a, b2b, summed, cnt, (float*)d_out, N_NODES / 16);
}
